// EdgeConditionedConvLayer_10385230921954
// MI455X (gfx1250) — matrix operand order repaired and checked by reading, not compiled, not run on hardware
//
#include <hip/hip_runtime.h>
#include <hip/hip_bf16.h>

typedef __bf16 bf16;
typedef __bf16 v16bf __attribute__((ext_vector_type(16)));
typedef float  v8f   __attribute__((ext_vector_type(8)));

#define N_NODES 10000
#define E_EDGES 250000
#define HD      256      // H
#define DE      64
#define DIN     320      // H + DE
#define TILE_E  64
#define EI_LD   328      // padded leading dim (bf16) for ei tile  (656B rows, 16B aligned, conflict-free)
#define HID_LD  264      // padded leading dim (bf16) for hidden   (528B rows, 16B aligned)

// ---------------------------------------------------------------------------
// fp32 -> bf16 bulk convert (vectorized: float4 in, 4xbf16 out)
// ---------------------------------------------------------------------------
__global__ __launch_bounds__(256) void cvt_f32_bf16_kernel(
    const float* __restrict__ src, bf16* __restrict__ dst, int n4) {
  int i = blockIdx.x * blockDim.x + threadIdx.x;
  if (i < n4) {
    float4 f = ((const float4*)src)[i];
    union { bf16 h[4]; uint2 u; } o;
    o.h[0] = (bf16)f.x; o.h[1] = (bf16)f.y;
    o.h[2] = (bf16)f.z; o.h[3] = (bf16)f.w;
    ((uint2*)dst)[i] = o.u;
  }
}

// ---------------------------------------------------------------------------
// Pack a row-major fp32 weight W[K x 256] into bf16 fragment-major layout:
// tile t = kt*16 + nt holds a 32x16 (KxN) tile; lane l owns 16 contiguous
// bf16: element e -> K = kt*32 + (e<8 ? (l>>4)*8 + e : 16 + (l>>4)*8 + e-8), N = nt*16 + (l&15).
// Matches the CDNA5 16-bit B-matrix 32x16 VGPR layout -> one 32B load/lane.
// ---------------------------------------------------------------------------
__global__ __launch_bounds__(256) void pack_weight_kernel(
    const float* __restrict__ W, bf16* __restrict__ P, int K) {
  int id = blockIdx.x * blockDim.x + threadIdx.x;
  int total = (K / 32) * 16 * 512;
  if (id >= total) return;
  int t  = id >> 9;
  int r  = id & 511;
  int l  = r >> 4;
  int e  = r & 15;
  int kt = t >> 4, nt = t & 15;
  int half = l >> 4, col = l & 15;
  int k = kt * 32 + ((e < 8) ? (half * 8 + e) : (16 + half * 8 + (e - 8)));
  int n = nt * 16 + col;
  P[id] = (bf16)W[k * HD + n];
}

__global__ __launch_bounds__(256) void zero_f32_kernel(float* __restrict__ p, int n) {
  int i = blockIdx.x * blockDim.x + threadIdx.x;
  if (i < n) p[i] = 0.0f;
}

// ---------------------------------------------------------------------------
// Main edge kernel. Per 64-edge tile:
//   stage ei = [x_bf[src] | ea_bf]  via GLOBAL_LOAD_ASYNC_TO_LDS_B128
//   hidden = relu(ei @ W1 + b1)     (WMMA bf16, staged in LDS)
//   gate   = sigmoid(ei @ Wg + bg)  (WMMA, f32 accum in registers)
//   core   = hidden @ W2 + b2       (WMMA, f32 accum in registers)
//   atomicAdd(agg[dst], gate * core)
// 8 waves; wave w computes M-tile (w&3), N-tiles [(w>>2)*8, +8).
// ---------------------------------------------------------------------------
__global__ __launch_bounds__(256) void edge_msg_kernel(
    const bf16*  __restrict__ x_bf,
    const bf16*  __restrict__ ea_bf,
    const int*   __restrict__ srcIdx,
    const int*   __restrict__ dstIdx,
    const bf16*  __restrict__ W1p,
    const bf16*  __restrict__ Wgp,
    const bf16*  __restrict__ W2p,
    const float* __restrict__ b1,
    const float* __restrict__ bg,
    const float* __restrict__ b2,
    float* __restrict__ agg) {
  extern __shared__ char smem_raw[];
  bf16* ei   = (bf16*)smem_raw;                       // TILE_E x EI_LD
  bf16* hid  = ei + TILE_E * EI_LD;                   // TILE_E x HID_LD
  int*  dstb = (int*)(hid + TILE_E * HID_LD);         // TILE_E

  const int tid  = threadIdx.x;
  const int e0   = blockIdx.x * TILE_E;
  const int w    = tid >> 5;
  const int lane = tid & 31;

  // --- stage dst indices (−1 marks padding edges) ---
  if (tid < TILE_E) {
    int e = e0 + tid;
    dstb[tid] = (e < E_EDGES) ? dstIdx[e] : -1;
  }

  // --- async-stage ei tile: wave w owns rows [w*8, w*8+8) ---
  // x part: one async B128 per row (32 lanes x 16B = 512B = one bf16 row)
#pragma unroll
  for (int i = 0; i < 8; ++i) {
    int m = w * 8 + i;
    int e = e0 + m;
    if (e < E_EDGES) {                       // wave-uniform branch
      int s = srcIdx[e];
      unsigned long long g =
          (unsigned long long)(uintptr_t)(x_bf + (size_t)s * HD) +
          (unsigned)(lane * 16);
      unsigned l = (unsigned)(uintptr_t)(ei + m * EI_LD) + (unsigned)(lane * 16);
      asm volatile("global_load_async_to_lds_b128 %0, %1, off"
                   :: "v"(l), "v"(g) : "memory");
    }
  }
  // edge_attr part: one async B128 covers 4 rows (8 lanes x 16B = 128B/row)
#pragma unroll
  for (int i = 0; i < 2; ++i) {
    int m = w * 8 + i * 4 + (lane >> 3);
    int e = e0 + m;
    if (e < E_EDGES) {                       // per-lane EXEC mask
      unsigned long long g =
          (unsigned long long)(uintptr_t)(ea_bf + (size_t)e * DE) +
          (unsigned)((lane & 7) * 16);
      unsigned l = (unsigned)(uintptr_t)(ei + m * EI_LD + HD) +
                   (unsigned)((lane & 7) * 16);
      asm volatile("global_load_async_to_lds_b128 %0, %1, off"
                   :: "v"(l), "v"(g) : "memory");
    }
  }
  asm volatile("s_wait_asynccnt 0x0" ::: "memory");
  __syncthreads();

  const int half = lane >> 4;     // which 16-lane group
  const int col  = lane & 15;     // N within tile (C layout)
  const int arow = lane & 15;     // M within tile (A layout)
  const int mt   = w & 3;
  const int ntb  = (w >> 2) * 8;
  const int m0   = mt * 16;

  union Frag { v16bf v; uint4 u[2]; };

  // --- Phase A: hidden tiles ---
  for (int nti = 0; nti < 8; ++nti) {
    int nt = ntb + nti;
    v8f acc = {};
#pragma unroll
    for (int kt = 0; kt < DIN / 32; ++kt) {
      Frag a, b;
      const bf16* ap = &ei[(m0 + arow) * EI_LD + kt * 32 + half * 8];
      a.u[0] = *(const uint4*)(ap);
      a.u[1] = *(const uint4*)(ap + 16);
      const uint4* bp = (const uint4*)(W1p + ((kt * 16 + nt) << 9) + lane * 16);
      b.u[0] = bp[0];
      b.u[1] = bp[1];
      acc = __builtin_amdgcn_wmma_f32_16x16x32_bf16(
          false, a.v, false, b.v, (short)0, acc, false, false);
    }
    float bias = b1[nt * 16 + col];
#pragma unroll
    for (int r = 0; r < 8; ++r) {
      float v = acc[r] + bias;
      v = v > 0.0f ? v : 0.0f;
      hid[(m0 + r + half * 8) * HID_LD + nt * 16 + col] = (bf16)v;
    }
  }
  __syncthreads();

  // --- Phase B/C fused per N-tile: gate GEMM + core GEMM + scatter ---
  for (int nti = 0; nti < 8; ++nti) {
    int nt = ntb + nti;
    v8f gacc = {};
#pragma unroll
    for (int kt = 0; kt < DIN / 32; ++kt) {
      Frag a, b;
      const bf16* ap = &ei[(m0 + arow) * EI_LD + kt * 32 + half * 8];
      a.u[0] = *(const uint4*)(ap);
      a.u[1] = *(const uint4*)(ap + 16);
      const uint4* bp = (const uint4*)(Wgp + ((kt * 16 + nt) << 9) + lane * 16);
      b.u[0] = bp[0];
      b.u[1] = bp[1];
      gacc = __builtin_amdgcn_wmma_f32_16x16x32_bf16(
          false, a.v, false, b.v, (short)0, gacc, false, false);
    }
    v8f cacc = {};
#pragma unroll
    for (int kt = 0; kt < HD / 32; ++kt) {
      Frag a, b;
      const bf16* ap = &hid[(m0 + arow) * HID_LD + kt * 32 + half * 8];
      a.u[0] = *(const uint4*)(ap);
      a.u[1] = *(const uint4*)(ap + 16);
      const uint4* bp = (const uint4*)(W2p + ((kt * 16 + nt) << 9) + lane * 16);
      b.u[0] = bp[0];
      b.u[1] = bp[1];
      cacc = __builtin_amdgcn_wmma_f32_16x16x32_bf16(
          false, a.v, false, b.v, (short)0, cacc, false, false);
    }
    float biasg = bg[nt * 16 + col];
    float bias2 = b2[nt * 16 + col];
#pragma unroll
    for (int r = 0; r < 8; ++r) {
      int m = m0 + r + half * 8;
      int d = dstb[m];
      if (d >= 0) {
        float g   = 1.0f / (1.0f + __expf(-(gacc[r] + biasg)));
        float msg = g * (cacc[r] + bias2);
        atomicAdd(&agg[d * HD + nt * 16 + col], msg);
      }
    }
  }
}

// ---------------------------------------------------------------------------
// LayerNorm: one wave per node row (H=256 -> 8 elems/lane, wave32 shfl).
// ---------------------------------------------------------------------------
__global__ __launch_bounds__(256) void layernorm_kernel(
    const float* __restrict__ agg, const float* __restrict__ x,
    const float* __restrict__ gamma, const float* __restrict__ beta,
    float* __restrict__ out) {
  int row  = blockIdx.x * 8 + (threadIdx.x >> 5);
  int lane = threadIdx.x & 31;
  if (row >= N_NODES) return;
  float v[8];
  float s = 0.0f, s2 = 0.0f;
#pragma unroll
  for (int j = 0; j < 8; ++j) {
    int c = lane + j * 32;
    float t = agg[row * HD + c] + x[row * HD + c];
    v[j] = t;
    s += t;
    s2 += t * t;
  }
#pragma unroll
  for (int o = 16; o > 0; o >>= 1) {
    s  += __shfl_xor(s, o, 32);
    s2 += __shfl_xor(s2, o, 32);
  }
  float mu   = s * (1.0f / HD);
  float var  = s2 * (1.0f / HD) - mu * mu;
  float rstd = rsqrtf(var + 1e-5f);
#pragma unroll
  for (int j = 0; j < 8; ++j) {
    int c = lane + j * 32;
    out[row * HD + c] = (v[j] - mu) * rstd * gamma[c] + beta[c];
  }
}

extern "C" void kernel_launch(void* const* d_in, const int* in_sizes, int n_in,
                              void* d_out, int out_size, void* d_ws, size_t ws_size,
                              hipStream_t stream) {
  (void)in_sizes; (void)n_in; (void)out_size; (void)ws_size;
  const float* x         = (const float*)d_in[0];
  const int*   edge_idx  = (const int*)d_in[1];   // [2, E]
  const float* edge_attr = (const float*)d_in[2];
  const float* W1        = (const float*)d_in[3];
  const float* b1        = (const float*)d_in[4];
  const float* W2        = (const float*)d_in[5];
  const float* b2        = (const float*)d_in[6];
  const float* Wg        = (const float*)d_in[7];
  const float* bg        = (const float*)d_in[8];
  const float* gamma     = (const float*)d_in[9];
  const float* beta      = (const float*)d_in[10];
  float* out = (float*)d_out;

  const int* srcIdx = edge_idx;
  const int* dstIdx = edge_idx + E_EDGES;

  // Workspace layout (bytes):
  //   agg   : N*H*4            = 10,240,000
  //   W1p   : 10*16*512*2      =    163,840
  //   Wgp   : 10*16*512*2      =    163,840
  //   W2p   :  8*16*512*2      =    131,072
  //   x_bf  : N*H*2            =  5,120,000
  //   ea_bf : E*DE*2           = 32,000,000
  char* wsp = (char*)d_ws;
  float* agg = (float*)wsp;                         wsp += (size_t)N_NODES * HD * sizeof(float);
  bf16*  W1p = (bf16*)wsp;                          wsp += (size_t)(DIN / 32) * 16 * 512 * sizeof(bf16);
  bf16*  Wgp = (bf16*)wsp;                          wsp += (size_t)(DIN / 32) * 16 * 512 * sizeof(bf16);
  bf16*  W2p = (bf16*)wsp;                          wsp += (size_t)(HD  / 32) * 16 * 512 * sizeof(bf16);
  bf16*  x_bf = (bf16*)wsp;                         wsp += (size_t)N_NODES * HD * sizeof(bf16);
  bf16*  ea_bf = (bf16*)wsp;

  // 1) zero aggregation buffer
  {
    int n = N_NODES * HD;
    zero_f32_kernel<<<(n + 255) / 256, 256, 0, stream>>>(agg, n);
  }
  // 2) bf16 conversions of gather sources
  {
    int n4 = N_NODES * HD / 4;
    cvt_f32_bf16_kernel<<<(n4 + 255) / 256, 256, 0, stream>>>(x, x_bf, n4);
  }
  {
    int n4 = E_EDGES * DE / 4;
    cvt_f32_bf16_kernel<<<(n4 + 255) / 256, 256, 0, stream>>>(edge_attr, ea_bf, n4);
  }
  // 3) pack weights to bf16 fragment-major
  pack_weight_kernel<<<((DIN / 32) * 16 * 512 + 255) / 256, 256, 0, stream>>>(W1, W1p, DIN);
  pack_weight_kernel<<<((DIN / 32) * 16 * 512 + 255) / 256, 256, 0, stream>>>(Wg, Wgp, DIN);
  pack_weight_kernel<<<((HD  / 32) * 16 * 512 + 255) / 256, 256, 0, stream>>>(W2, W2p, HD);

  // 4) fused edge MLP + gate + scatter
  {
    size_t smem = (size_t)(TILE_E * EI_LD + TILE_E * HID_LD) * sizeof(bf16)
                + TILE_E * sizeof(int);                // 76,032 B (< 320KB LDS/WGP)
    int grid = (E_EDGES + TILE_E - 1) / TILE_E;        // 3907
    edge_msg_kernel<<<grid, 256, smem, stream>>>(
        x_bf, ea_bf, srcIdx, dstIdx, W1p, Wgp, W2p, b1, bg, b2, agg);
  }
  // 5) residual + layernorm
  layernorm_kernel<<<(N_NODES + 7) / 8, 256, 0, stream>>>(agg, x, gamma, beta, out);
}
